// PSConvNetBase_85667417686617
// MI455X (gfx1250) — matrix operand order repaired and checked by reading, not compiled, not run on hardware
//
#include <hip/hip_runtime.h>

// Locally-connected 1D conv for MI455X (gfx1250, wave32, WMMA).
// Per seq position s: [64 x 448] x [448 x 128] GEMM + bias + ReLU.
// f32 -> f16 conversion at LDS staging, v_wmma_f32_16x16x32_f16 with f32 accum.
// One workgroup (8 waves / 256 threads) per s; K double-buffered in LDS.
// W reads / out writes use non-temporal cache policy (stream-once data);
// input x stays temporal (L2-resident, reused by 7 neighboring blocks).
// Main loop force-unrolled; all B fragments loaded before the WMMA burst so
// ds_load latency is overlapped (partial dscnt waits, not dscnt==0 per tile).

typedef _Float16 half2t __attribute__((ext_vector_type(2)));
typedef _Float16 half8  __attribute__((ext_vector_type(8)));
typedef _Float16 v16h   __attribute__((ext_vector_type(16)));
typedef float    v4f    __attribute__((ext_vector_type(4)));
typedef float    v8f    __attribute__((ext_vector_type(8)));

#define BATCH 64
#define FEAT  64
#define SEQL  1024
#define WINW  7
#define PADW  3
#define KTOT  448          // WIN * FEAT
#define FILT  128
#define KC    32           // K per chunk
#define NCH   (KTOT / KC)  // 14 chunks
#define KSTR  40           // 32 halves + 8 pad: 16B-aligned rows, bank-conflict-free

__device__ __forceinline__ void stage_chunk(
    int c, _Float16* __restrict__ sA, _Float16* __restrict__ sB,
    const float* __restrict__ x, const float* __restrict__ Wc,
    int s, int tid)
{
    // ---- W chunk: [32 k][128 n] f32. Each thread loads the same 4 n's from two
    // adjacent k-rows (coalesced 16B, non-temporal: W is streamed once),
    // packs (k,k+1) half pairs, and stores transposed [n][k] with ds_store_b32.
    const float* Wk = Wc + (size_t)c * KC * FILT;
#pragma unroll
    for (int i = 0; i < 2; ++i) {
        int fl = tid + i * 256;        // 0..511
        int kp = (fl >> 5) << 1;       // even k row: 0,2,...,30
        int n  = (fl & 31) << 2;       // 0,4,...,124
        v4f wa = __builtin_nontemporal_load((const v4f*)(Wk + (size_t)kp * FILT + n));
        v4f wb = __builtin_nontemporal_load((const v4f*)(Wk + (size_t)(kp + 1) * FILT + n));
        half2t p0 = {(_Float16)wa.x, (_Float16)wb.x};
        half2t p1 = {(_Float16)wa.y, (_Float16)wb.y};
        half2t p2 = {(_Float16)wa.z, (_Float16)wb.z};
        half2t p3 = {(_Float16)wa.w, (_Float16)wb.w};
        *(half2t*)(sB + (n + 0) * KSTR + kp) = p0;
        *(half2t*)(sB + (n + 1) * KSTR + kp) = p1;
        *(half2t*)(sB + (n + 2) * KSTR + kp) = p2;
        *(half2t*)(sB + (n + 3) * KSTR + kp) = p3;
    }
    // ---- A chunk: k in [c*32, c*32+32) => window w = c>>1, feats fb..fb+31,
    // pos = s + w - PAD, zero-padded outside [0,SEQ). Branchless: clamp the
    // (uniform) position so the load is always in-bounds, scale by 0/1 mask.
    // Two adjacent feats per thread, packed into one conflict-free b32 store.
    {
        const int w    = c >> 1;
        const int fb   = (c & 1) << 5;
        const int pos  = s + w - PADW;
        const int posc = pos < 0 ? 0 : (pos > SEQL - 1 ? SEQL - 1 : pos);
        const float msk = (pos >= 0 && pos < SEQL) ? 1.0f : 0.0f;
#pragma unroll
        for (int i = 0; i < 4; ++i) {
            int e  = tid + i * 256;    // 0..1023
            int b  = e >> 4;           // batch 0..63
            int k2 = (e & 15) << 1;    // 0,2,...,30
            const float* xp = x + ((size_t)b * FEAT + fb + k2) * SEQL + posc;
            float v0 = xp[0]    * msk;
            float v1 = xp[SEQL] * msk;
            half2t p = {(_Float16)v0, (_Float16)v1};
            *(half2t*)(sA + b * KSTR + k2) = p;
        }
    }
}

__global__ void __launch_bounds__(256)
lc1d_wmma_kernel(const float* __restrict__ x,     // [64][64][1024]
                 const float* __restrict__ W,     // [1024][448][128]
                 const float* __restrict__ bias,  // [1024][128]
                 float* __restrict__ out)         // [64][1024][128]
{
    __shared__ __align__(16) _Float16 sA[2][BATCH * KSTR];   // 10 KB
    __shared__ __align__(16) _Float16 sB[2][FILT * KSTR];    // 20 KB

    const int s    = blockIdx.x;
    const int tid  = threadIdx.x;
    const int lane = tid & 31;
    const int wv   = tid >> 5;             // wave 0..7
    const int mB   = (wv & 3) * 16;        // batch tile base
    const int nB   = (wv >> 2) * 64;       // filter group base (4 tiles of 16)
    const int lrow = lane & 15;
    const int kSel = lane >> 4;            // 0 / 1 (lane half)

    const float* Wc = W + (size_t)s * KTOT * FILT;

    // Preload bias into accumulators: C[m][n] = b_all[s][n] (broadcast over M).
    v8f acc[4];
#pragma unroll
    for (int t = 0; t < 4; ++t) {
        float bv = bias[(size_t)s * FILT + nB + t * 16 + lrow];
        acc[t] = (v8f){bv, bv, bv, bv, bv, bv, bv, bv};
    }

    stage_chunk(0, sA[0], sB[0], x, Wc, s, tid);
    __syncthreads();

#pragma unroll
    for (int c = 0; c < NCH; ++c) {
        const int cur = c & 1;
        if (c + 1 < NCH)
            stage_chunk(c + 1, sA[cur ^ 1], sB[cur ^ 1], x, Wc, s, tid);

        // A fragment (16-bit A 16x32 layout): lanes 0-15 K{0..7,16..23},
        // lanes 16-31 K{8..15,24..31}, 8 consecutive halves per ds_load_b128.
        const _Float16* aP = &sA[cur][(mB + lrow) * KSTR];
        half8 a0 = *(const half8*)(aP + kSel * 8);
        half8 a1 = *(const half8*)(aP + 16 + kSel * 8);
        v16h av  = __builtin_shufflevector(a0, a1,
                     0,1,2,3,4,5,6,7,8,9,10,11,12,13,14,15);

        // Load ALL B fragments first so the 8 ds_load_b128 batch into a clause
        // and WMMAs only take partial dscnt waits.
        v16h bfrag[4];
#pragma unroll
        for (int t = 0; t < 4; ++t) {
            // B fragment: lanes 0-15 hold K{0..7,16..23}, lanes 16-31 K{8..15,24..31}
            // (same order as A), column n = lane&15 -> transposed [n][k] LDS layout.
            const _Float16* bP =
                &sB[cur][(nB + t * 16 + lrow) * KSTR + kSel * 8];
            half8 b0 = *(const half8*)(bP);
            half8 b1 = *(const half8*)(bP + 16);
            bfrag[t] = __builtin_shufflevector(b0, b1,
                         0,1,2,3,4,5,6,7,8,9,10,11,12,13,14,15);
        }
#pragma unroll
        for (int t = 0; t < 4; ++t) {
            acc[t] = __builtin_amdgcn_wmma_f32_16x16x32_f16(
                false, av, false, bfrag[t], (short)0, acc[t], false, false);
        }
        __syncthreads();
    }

    // C/D layout: VGPR r -> M = r + 8*(lane>>4); N = lane&15. Fused ReLU.
    // Output written once, never re-read on device -> non-temporal stores.
#pragma unroll
    for (int t = 0; t < 4; ++t) {
        const int n = nB + t * 16 + lrow;
#pragma unroll
        for (int r = 0; r < 8; ++r) {
            const int b = mB + r + kSel * 8;
            __builtin_nontemporal_store(fmaxf(acc[t][r], 0.0f),
                                        &out[((size_t)b * SEQL + s) * FILT + n]);
        }
    }
}

extern "C" void kernel_launch(void* const* d_in, const int* in_sizes, int n_in,
                              void* d_out, int out_size, void* d_ws, size_t ws_size,
                              hipStream_t stream) {
    const float* x    = (const float*)d_in[0];   // input_mat [64][64][1024]
    const float* Wall = (const float*)d_in[1];   // W_all [1024][448][128]
    const float* ball = (const float*)d_in[2];   // b_all [1024][128]
    float* out        = (float*)d_out;           // [64][1024][128]
    (void)in_sizes; (void)n_in; (void)out_size; (void)d_ws; (void)ws_size;

    lc1d_wmma_kernel<<<SEQL, 256, 0, stream>>>(x, Wall, ball, out);
}
